// ODERNNmodel_64364379898426
// MI455X (gfx1250) — matrix operand order repaired and checked by reading, not compiled, not run on hardware
//
#include <hip/hip_runtime.h>

// ---------------------------------------------------------------------------
// ODE-RNN on MI455X (gfx1250), wave32 + WMMA bf16.
//
// B=128, F=512, I=128, H=512, O=1, NSTEP=5.
// Grid: 8 persistent workgroups, one per 16-row batch block (independent).
// Block: 512 threads = 16 wave32; each wave owns 2 N-tiles (32 tiles = 512 cols).
// Weights pre-swizzled to WMMA B-fragment layout (bf16) in d_ws by pack kernel;
// they stay L2-hot (1.7 MB total vs 192 MB L2) and are streamed per k-block.
// h (f32) and time-mean accumulator live in registers (lane owns its C-tile
// coords); bf16 copies of h / relu-activation / x live in LDS (padded stride).
//
// Round-2 fix: k-loop must remain rolled. Full unroll made B-fragment loads
// loop-invariant across the time/substep loops -> LICM hoisted ~256 VGPRs of
// weights -> scratch spills/reloads in the inner loop. unroll(disable) keeps
// the loop body at 2x ds_load_b128 + 4x global_load_b128 + 2x wmma.
// ---------------------------------------------------------------------------

typedef __attribute__((ext_vector_type(16))) __bf16 v16bf;
typedef __attribute__((ext_vector_type(8)))  __bf16 v8bf;
typedef __attribute__((ext_vector_type(8)))  float  v8f;

#define B_   128
#define F_   512
#define I_   128
#define H_   512
#define NSTEP_ 5
#define HS   520   // padded LDS row stride for H-wide bf16 rows (16B aligned, bank-skewed)
#define XS   136   // padded LDS row stride for I-wide bf16 rows

__device__ __forceinline__ unsigned short f2bf(float f) {
    unsigned u = __float_as_uint(f);
    u += 0x7FFFu + ((u >> 16) & 1u);   // round-to-nearest-even
    return (unsigned short)(u >> 16);
}

// ---------------------------------------------------------------------------
// Pack weight W[N][K] (row-major f32, computing y = act @ W^T) into the
// per-lane bf16 B-fragment layout consumed by v_wmma_f32_16x16x32_bf16:
//   frag(tile_n, k_blk): 32 lanes x 16 bf16, lane l -> column N = tile_n*16 + (l&15),
//   element j -> K = k_blk*32 + 16*(j>>3) + 8*(l>>4) + (j&7).
// Linear index: idx = ((tile_n*(K/32) + k_blk)*32 + lane)*16 + j
// ---------------------------------------------------------------------------
__global__ void pack_weight_kernel(const float* __restrict__ W,
                                   unsigned short* __restrict__ out,
                                   int N, int K) {
    int idx = blockIdx.x * blockDim.x + threadIdx.x;
    if (idx >= N * K) return;
    int j    = idx & 15;
    int ln   = (idx >> 4) & 31;
    int rest = idx >> 9;
    int kb   = K >> 5;
    int kblk = rest % kb;
    int ntl  = rest / kb;
    int n = ntl * 16 + (ln & 15);
    int k = kblk * 32 + ((j >> 3) << 4) + ((ln >> 4) << 3) + (j & 7);
    out[idx] = f2bf(W[n * K + k]);
}

// ---------------------------------------------------------------------------
// 2-tile GEMM micro-kernel: acc[t] (+)= A[16 x K] @ Bpack(tile0+t)  (bf16->f32)
// A in LDS (bf16 as ushort, row stride strideA elems). A-fragment per ISA:
// lane l: M = l&15, K-half = l>>4; elems 0..7 -> K=k0+8*hi.., elems 8..15 ->
// K=k0+16+8*hi..  (two 16B ds loads). B-fragment: one 32B coalesced load.
// ---------------------------------------------------------------------------
__device__ __forceinline__ void gemm2(const unsigned short* __restrict__ A,
                                      int strideA, int K,
                                      const unsigned short* __restrict__ Bpack,
                                      int tile0, int lane, v8f acc[2]) {
    const int kb = K >> 5;
    const int hi = lane >> 4;
    const int m  = lane & 15;
    const unsigned short* arow = A + m * strideA;
    const v16bf* Bp0 = (const v16bf*)Bpack + (size_t)tile0 * kb + lane * 0; // base
#pragma clang loop unroll(disable)
    for (int kblk = 0; kblk < kb; ++kblk) {
        const int k0 = kblk << 5;
        v8bf alo = *(const v8bf*)(arow + k0 + 8 * hi);
        v8bf ahi = *(const v8bf*)(arow + k0 + 16 + 8 * hi);
        v16bf af = __builtin_shufflevector(alo, ahi,
                      0,1,2,3,4,5,6,7,8,9,10,11,12,13,14,15);
        // prefetch next k-block of tile0's B stream (global_prefetch_b8)
        __builtin_prefetch((const void*)&Bp0[(size_t)(kblk + 1) * 32 + lane], 0, 1);
#pragma unroll
        for (int t = 0; t < 2; ++t) {
            v16bf bf = Bp0[((size_t)t * kb + kblk) * 32 + lane];
            acc[t] = __builtin_amdgcn_wmma_f32_16x16x32_bf16(
                         false, af, false, bf, (short)0, acc[t], false, false);
        }
    }
}

// ---------------------------------------------------------------------------
// Main persistent kernel: one workgroup per 16 batch rows, all F steps.
// ---------------------------------------------------------------------------
__global__ __launch_bounds__(512, 1)
void odernn_kernel(const float* __restrict__ x,   const float* __restrict__ tp,
                   const float* __restrict__ b1,  const float* __restrict__ b2,
                   const float* __restrict__ bih, const float* __restrict__ bhh,
                   const float* __restrict__ Wc,  const float* __restrict__ bc,
                   const unsigned short* __restrict__ W1p,
                   const unsigned short* __restrict__ W2p,
                   const unsigned short* __restrict__ Whhp,
                   const unsigned short* __restrict__ Wihp,
                   float* __restrict__ out) {
    __shared__ __align__(16) unsigned short hbf[16 * HS];  // bf16 mirror of h
    __shared__ __align__(16) unsigned short abf[16 * HS];  // relu activation
    __shared__ __align__(16) unsigned short xbf[16 * XS];  // x_t slice
    __shared__ float scaleL[16];
    __shared__ float b1L[H_], b2L[H_], brnL[H_];
    __shared__ float zrow[16];

    const int tid  = threadIdx.x;
    const int lane = tid & 31;
    const int wv   = tid >> 5;
    const int rb   = blockIdx.x * 16;          // first batch row of this block
    const int tile0 = wv * 2;                  // 2 N-tiles per wave
    const int hi  = lane >> 4;
    const int col = lane & 15;

    // Register-resident state: h and running time-sum, one (t,v) coord each:
    //   M = v + 8*hi, N = (tile0+t)*16 + col
    float hreg[2][8];
    float sreg[2][8];
#pragma unroll
    for (int t = 0; t < 2; ++t)
#pragma unroll
        for (int v = 0; v < 8; ++v) { hreg[t][v] = 0.f; sreg[t][v] = 0.f; }

    for (int i = tid; i < 16 * HS; i += 512) hbf[i] = 0;      // h0 = 0
    for (int i = tid; i < H_; i += 512) {
        b1L[i]  = b1[i];
        b2L[i]  = b2[i];
        brnL[i] = bih[i] + bhh[i];
    }
    if (tid < 16) zrow[tid] = 0.f;
    __syncthreads();

#pragma clang loop unroll(disable)
    for (int f = 0; f < F_; ++f) {
        // --- per-row Euler scale + x_t staging -----------------------------
        if (tid < 16) {
            float d = 0.f;
            if (f > 0) d = tp[(rb + tid) * F_ + f] - tp[(rb + tid) * F_ + f - 1];
            scaleL[tid] = d * (1.0f / NSTEP_);
        }
        for (int e = tid; e < 16 * I_; e += 512) {
            int r = e >> 7, k = e & (I_ - 1);
            xbf[r * XS + k] =
                f2bf(x[((size_t)(rb + r) * F_ + f) * I_ + k]);
        }
        __syncthreads();

        // --- ODE: 5 explicit-Euler substeps --------------------------------
#pragma clang loop unroll(disable)
        for (int s = 0; s < NSTEP_; ++s) {
            // stage 1: a = relu(h @ W1^T + b1)
            v8f acc[2] = {{0,0,0,0,0,0,0,0},{0,0,0,0,0,0,0,0}};
            gemm2(hbf, HS, H_, W1p, tile0, lane, acc);
#pragma unroll
            for (int t = 0; t < 2; ++t) {
                int n = (tile0 + t) * 16 + col;
#pragma unroll
                for (int v = 0; v < 8; ++v) {
                    int m = v + 8 * hi;
                    float a = acc[t][v] + b1L[n];
                    abf[m * HS + n] = f2bf(a > 0.f ? a : 0.f);
                }
            }
            __syncthreads();   // abf ready for stage 2

            // stage 2: h += (a @ W2^T + b2) * (dt/NSTEP)
            v8f acc2[2] = {{0,0,0,0,0,0,0,0},{0,0,0,0,0,0,0,0}};
            gemm2(abf, HS, H_, W2p, tile0, lane, acc2);
#pragma unroll
            for (int t = 0; t < 2; ++t) {
                int n = (tile0 + t) * 16 + col;
#pragma unroll
                for (int v = 0; v < 8; ++v) {
                    int m = v + 8 * hi;
                    hreg[t][v] += (acc2[t][v] + b2L[n]) * scaleL[m];
                    hbf[m * HS + n] = f2bf(hreg[t][v]);
                }
            }
            __syncthreads();   // hbf ready for next substep / RNN
        }

        // --- RNN cell: h = tanh(x@Wih^T + h@Whh^T + bih + bhh) -------------
        v8f acc3[2] = {{0,0,0,0,0,0,0,0},{0,0,0,0,0,0,0,0}};
        gemm2(xbf, XS, I_, Wihp, tile0, lane, acc3);
        gemm2(hbf, HS, H_, Whhp, tile0, lane, acc3);
        __syncthreads();       // everyone done reading hbf before overwrite
#pragma unroll
        for (int t = 0; t < 2; ++t) {
            int n = (tile0 + t) * 16 + col;
#pragma unroll
            for (int v = 0; v < 8; ++v) {
                int m = v + 8 * hi;
                float hn = tanhf(acc3[t][v] + brnL[n]);
                hreg[t][v] = hn;
                sreg[t][v] += hn;
                hbf[m * HS + n] = f2bf(hn);
            }
        }
        __syncthreads();
    }

    // --- classifier: out = sigmoid(mean_t(h) @ Wc^T + bc), O == 1 ----------
    const float invF = 1.0f / (float)F_;
#pragma unroll
    for (int v = 0; v < 8; ++v) {
        int m = v + 8 * hi;
        float p = 0.f;
#pragma unroll
        for (int t = 0; t < 2; ++t) {
            int n = (tile0 + t) * 16 + col;
            p += sreg[t][v] * Wc[n];
        }
        atomicAdd(&zrow[m], p * invF);   // ds_add_f32
    }
    __syncthreads();
    if (tid < 16) {
        float z = zrow[tid] + bc[0];
        out[rb + tid] = 1.0f / (1.0f + __expf(-z));
    }
}

// ---------------------------------------------------------------------------
extern "C" void kernel_launch(void* const* d_in, const int* in_sizes, int n_in,
                              void* d_out, int out_size, void* d_ws, size_t ws_size,
                              hipStream_t stream) {
    const float* x   = (const float*)d_in[0];
    const float* tp  = (const float*)d_in[1];
    const float* W1  = (const float*)d_in[2];
    const float* b1  = (const float*)d_in[3];
    const float* W2  = (const float*)d_in[4];
    const float* b2  = (const float*)d_in[5];
    const float* Wih = (const float*)d_in[6];
    const float* Whh = (const float*)d_in[7];
    const float* bih = (const float*)d_in[8];
    const float* bhh = (const float*)d_in[9];
    const float* Wc  = (const float*)d_in[10];
    const float* bc  = (const float*)d_in[11];
    float* out = (float*)d_out;

    // Scratch layout (bf16 packed weights): needs ~1.7 MB.
    unsigned short* ws   = (unsigned short*)d_ws;
    unsigned short* W1p  = ws;
    unsigned short* W2p  = ws + (size_t)H_ * H_;
    unsigned short* Whhp = ws + (size_t)2 * H_ * H_;
    unsigned short* Wihp = ws + (size_t)3 * H_ * H_;

    const int P = 256;
    pack_weight_kernel<<<(H_ * H_ + P - 1) / P, P, 0, stream>>>(W1,  W1p,  H_, H_);
    pack_weight_kernel<<<(H_ * H_ + P - 1) / P, P, 0, stream>>>(W2,  W2p,  H_, H_);
    pack_weight_kernel<<<(H_ * H_ + P - 1) / P, P, 0, stream>>>(Whh, Whhp, H_, H_);
    pack_weight_kernel<<<(H_ * I_ + P - 1) / P, P, 0, stream>>>(Wih, Wihp, H_, I_);

    odernn_kernel<<<B_ / 16, 512, 0, stream>>>(
        x, tp, b1, b2, bih, bhh, Wc, bc, W1p, W2p, Whhp, Wihp, out);
}
